// SelfAttention_13958643712726
// MI455X (gfx1250) — matrix operand order repaired and checked by reading, not compiled, not run on hardware
//
#include <hip/hip_runtime.h>
#include <hip/hip_bf16.h>

// ---------------------------------------------------------------------------
// Types for CDNA5 WMMA (gfx1250, wave32)
// ---------------------------------------------------------------------------
typedef __bf16 bf16;
typedef __attribute__((ext_vector_type(16))) __bf16 v16bf;
typedef __attribute__((ext_vector_type(8)))  __bf16 v8bf;
typedef __attribute__((ext_vector_type(4)))  __bf16 v4bf;
typedef __attribute__((ext_vector_type(8)))  float  v8f;

union ABu { v16bf v; v8bf h[2]; };

static __device__ __forceinline__ v8f wmma_bf16(v16bf a, v16bf b, v8f c) {
    // D(16x16 f32) = A(16x32 bf16) * B(32x16 bf16) + C
    return __builtin_amdgcn_wmma_f32_16x16x32_bf16(
        /*neg_a=*/false, a, /*neg_b=*/false, b,
        /*c_mod=*/(short)0, c, /*reuse_a=*/false, /*reuse_b=*/false);
}

static __device__ __forceinline__ v8bf cvt8(const float4& a, const float4& b) {
    v8bf r;
    r[0] = (bf16)a.x; r[1] = (bf16)a.y; r[2] = (bf16)a.z; r[3] = (bf16)a.w;
    r[4] = (bf16)b.x; r[5] = (bf16)b.y; r[6] = (bf16)b.z; r[7] = (bf16)b.w;
    return r;
}

// Problem constants
#define BB   4
#define NN   8
#define TT   512
#define DD   512
#define HH   8
#define HD   64
#define GG   4
#define DG   128           // DD / GG
#define ME   16384         // BB*NN*TT rows
#define NTR  4096          // NN*TT rows per batch
#define GNCH 32            // groupnorm row-chunks per (b,g)

// ---------------------------------------------------------------------------
// Weight f32 -> bf16 conversion (512x512 source, arbitrary dst stride/offset)
// ---------------------------------------------------------------------------
__global__ __launch_bounds__(256) void cvt_w_kernel(
    const float* __restrict__ src, bf16* __restrict__ dst,
    int dstStride, int colOff)
{
    size_t i4   = (size_t)blockIdx.x * 256 + threadIdx.x;   // 65536 float4s
    size_t base = i4 * 4;
    int r = (int)(base >> 9);
    int c = (int)(base & 511);
    float4 x = *(const float4*)(src + base);
    v4bf y;
    y[0] = (bf16)x.x; y[1] = (bf16)x.y; y[2] = (bf16)x.z; y[3] = (bf16)x.w;
    *(v4bf*)(dst + (size_t)r * dstStride + colOff + c) = y;
}

// ---------------------------------------------------------------------------
// GroupNorm: deterministic 2-stage reduction + apply (f32 in, bf16 out)
// X layout: (BB, 4096 rows, 512 cols); group g = cols [g*128, g*128+128)
// ---------------------------------------------------------------------------
__global__ __launch_bounds__(256) void gn_part_kernel(
    const float* __restrict__ X, float* __restrict__ part)
{
    int bg = blockIdx.y;            // 0..15  (b*4+g)
    int b  = bg >> 2, g = bg & 3;
    int chunk = blockIdx.x;         // 0..31
    const float* base = X + ((size_t)b * NTR + (size_t)chunk * 128) * DD + g * DG;
    float s = 0.f, s2 = 0.f;
    for (int idx = threadIdx.x; idx < 128 * DG; idx += 256) {
        int r = idx >> 7, c = idx & 127;
        float v = base[(size_t)r * DD + c];
        s += v; s2 += v * v;
    }
    __shared__ float sh0[256], sh1[256];
    sh0[threadIdx.x] = s; sh1[threadIdx.x] = s2;
    __syncthreads();
    for (int o = 128; o > 0; o >>= 1) {
        if ((int)threadIdx.x < o) {
            sh0[threadIdx.x] += sh0[threadIdx.x + o];
            sh1[threadIdx.x] += sh1[threadIdx.x + o];
        }
        __syncthreads();
    }
    if (threadIdx.x == 0) {
        part[(size_t)(bg * GNCH + chunk) * 2 + 0] = sh0[0];
        part[(size_t)(bg * GNCH + chunk) * 2 + 1] = sh1[0];
    }
}

__global__ void gn_stat_kernel(const float* __restrict__ part,
                               float* __restrict__ stats)
{
    int bg = blockIdx.x;            // 16 blocks, 32 threads
    int lane = threadIdx.x;
    float s  = part[(size_t)(bg * GNCH + lane) * 2 + 0];
    float s2 = part[(size_t)(bg * GNCH + lane) * 2 + 1];
    for (int o = 16; o > 0; o >>= 1) {
        s  += __shfl_xor(s,  o, 32);
        s2 += __shfl_xor(s2, o, 32);
    }
    if (lane == 0) {
        const float invN = 1.0f / (float)(NTR * DG);   // 524288 elements
        float mu  = s * invN;
        float var = s2 * invN - mu * mu;
        stats[bg * 2 + 0] = mu;
        stats[bg * 2 + 1] = rsqrtf(var + 1e-3f);
    }
}

__global__ __launch_bounds__(256) void gn_apply_kernel(
    const float* __restrict__ X, const float* __restrict__ stats,
    bf16* __restrict__ Y)
{
    size_t i4   = (size_t)blockIdx.x * 256 + threadIdx.x;  // 2,097,152 float4s
    size_t base = i4 * 4;
    int m = (int)(base >> 9);
    int c = (int)(base & 511);
    int b = m >> 12;                // / 4096
    int g = c >> 7;
    float mu = stats[(b * GG + g) * 2 + 0];
    float rs = stats[(b * GG + g) * 2 + 1];
    float4 x = *(const float4*)(X + base);
    v4bf y;
    y[0] = (bf16)((x.x - mu) * rs);
    y[1] = (bf16)((x.y - mu) * rs);
    y[2] = (bf16)((x.z - mu) * rs);
    y[3] = (bf16)((x.w - mu) * rs);
    *(v4bf*)(Y + base) = y;
}

// ---------------------------------------------------------------------------
// GEMM: out[m,n] = sum_k A[m,k] * W[n,k]   (A bf16 MxK, W bf16 NxK row-major)
// block = 256 threads = 8 waves; wave computes 16(M) x 64(N); grid(M/128, N/64)
// EPI 0: store f32 out[m*512+n]                 (Q0 / K0)
// EPI 1: store bf16 transposed per-head V  -> Vt[(bnn*8+h)*64+hd][t]
// EPI 2: bias + leaky_relu -> bf16 X3[m*1024 + 512 + n]   (FF hidden)
// EPI 3: + bias + bias2 + resid -> f32 out[m*512+n]       (final)
// ---------------------------------------------------------------------------
template<int EPI>
__global__ __launch_bounds__(256) void gemm_kernel(
    const bf16* __restrict__ A, const bf16* __restrict__ W,
    int K, int lda,
    float* __restrict__ outf, bf16* __restrict__ outb,
    const float* __restrict__ bias, const float* __restrict__ bias2,
    const float* __restrict__ resid)
{
    const int lane = threadIdx.x & 31;
    const int wave = threadIdx.x >> 5;
    const int half = lane >> 4;
    const int lr   = lane & 15;
    const int m0 = blockIdx.x * 128 + wave * 16;
    const int n0 = blockIdx.y * 64;

    v8f acc[4] = {};
    const bf16* ap = A + (size_t)(m0 + lr) * lda + half * 16;

    for (int k0 = 0; k0 < K; k0 += 32) {
        ABu a;
        a.h[0] = *(const v8bf*)(ap + k0);
        a.h[1] = *(const v8bf*)(ap + k0 + 8);
#pragma unroll
        for (int nt = 0; nt < 4; ++nt) {
            const bf16* bp = W + (size_t)(n0 + nt * 16 + lr) * K + k0 + half * 16;
            v16bf bv = *(const v16bf*)bp;
            acc[nt] = wmma_bf16(a.v, bv, acc[nt]);
        }
    }

#pragma unroll
    for (int nt = 0; nt < 4; ++nt) {
        const int n = n0 + nt * 16 + lr;
        if constexpr (EPI == 0) {
#pragma unroll
            for (int r = 0; r < 8; ++r) {
                int m = m0 + half * 8 + r;
                outf[(size_t)m * DD + n] = acc[nt][r];
            }
        } else if constexpr (EPI == 1) {
            int bnn = m0 >> 9;
            int tb  = (m0 & 511) + half * 8;
            int h = n >> 6, hd = n & 63;
            v8bf pk;
#pragma unroll
            for (int r = 0; r < 8; ++r) pk[r] = (bf16)acc[nt][r];
            *(v8bf*)(outb + ((size_t)(bnn * HH + h) * HD + hd) * TT + tb) = pk;
        } else if constexpr (EPI == 2) {
            float bv = bias[n];
#pragma unroll
            for (int r = 0; r < 8; ++r) {
                int m = m0 + half * 8 + r;
                float v = acc[nt][r] + bv;
                v = v > 0.f ? v : 0.01f * v;
                outb[(size_t)m * 1024 + 512 + n] = (bf16)v;
            }
        } else {
            float bv = bias[n] + bias2[n];
#pragma unroll
            for (int r = 0; r < 8; ++r) {
                int m = m0 + half * 8 + r;
                size_t off = (size_t)m * DD + n;
                outf[off] = acc[nt][r] + bv + resid[off];
            }
        }
    }
}

// ---------------------------------------------------------------------------
// Attention logits: S[bnh, q, key] = 0.125 * sum_hd Q[q,hd] * K[key,hd]
// grid(32 qtiles, 256 bnh); wave handles 4 key-tiles (n0 = wave*64)
// ---------------------------------------------------------------------------
__global__ __launch_bounds__(256) void attn_logits_kernel(
    const bf16* __restrict__ Qb, const bf16* __restrict__ Kb,
    float* __restrict__ S)
{
    const int lane = threadIdx.x & 31;
    const int wave = threadIdx.x >> 5;
    const int half = lane >> 4;
    const int lr   = lane & 15;
    const int qt  = blockIdx.x;
    const int bnh = blockIdx.y;
    const int bnn = bnh >> 3, h = bnh & 7;
    const int q0 = qt * 16;
    const int n0 = wave * 64;

    const bf16* qbase = Qb + ((size_t)bnn * TT + q0 + lr) * DD + h * HD + half * 16;
    const bf16* kbase = Kb + (size_t)bnn * TT * DD + h * HD + half * 16;

    v8f acc[4] = {};
#pragma unroll
    for (int k0 = 0; k0 < HD; k0 += 32) {
        ABu a;
        a.h[0] = *(const v8bf*)(qbase + k0);
        a.h[1] = *(const v8bf*)(qbase + k0 + 8);
#pragma unroll
        for (int nt = 0; nt < 4; ++nt) {
            const bf16* bp = kbase + (size_t)(n0 + nt * 16 + lr) * DD + k0;
            v16bf bv = *(const v16bf*)bp;
            acc[nt] = wmma_bf16(a.v, bv, acc[nt]);
        }
    }

#pragma unroll
    for (int nt = 0; nt < 4; ++nt) {
        int key = n0 + nt * 16 + lr;
#pragma unroll
        for (int r = 0; r < 8; ++r) {
            int q = q0 + half * 8 + r;
            S[((size_t)bnh * TT + q) * TT + key] = 0.125f * acc[nt][r];
        }
    }
}

// ---------------------------------------------------------------------------
// Row softmax over 512 elements, in place. One wave per row (16 elems/lane).
// ---------------------------------------------------------------------------
__global__ __launch_bounds__(256) void softmax_kernel(float* __restrict__ S)
{
    const int lane = threadIdx.x & 31;
    const int wave = threadIdx.x >> 5;
    size_t row = (size_t)blockIdx.x * 8 + wave;
    float* p = S + row * TT;

    float4 v[4];
#pragma unroll
    for (int j = 0; j < 4; ++j) v[j] = ((const float4*)p)[lane + 32 * j];

    float mx = -3.402823466e38f;
#pragma unroll
    for (int j = 0; j < 4; ++j) {
        mx = fmaxf(mx, fmaxf(fmaxf(v[j].x, v[j].y), fmaxf(v[j].z, v[j].w)));
    }
    for (int o = 16; o > 0; o >>= 1) mx = fmaxf(mx, __shfl_xor(mx, o, 32));

    float s = 0.f;
#pragma unroll
    for (int j = 0; j < 4; ++j) {
        v[j].x = __expf(v[j].x - mx); v[j].y = __expf(v[j].y - mx);
        v[j].z = __expf(v[j].z - mx); v[j].w = __expf(v[j].w - mx);
        s += v[j].x + v[j].y + v[j].z + v[j].w;
    }
    for (int o = 16; o > 0; o >>= 1) s += __shfl_xor(s, o, 32);
    float inv = 1.0f / s;

#pragma unroll
    for (int j = 0; j < 4; ++j) {
        v[j].x *= inv; v[j].y *= inv; v[j].z *= inv; v[j].w *= inv;
        ((float4*)p)[lane + 32 * j] = v[j];
    }
}

// ---------------------------------------------------------------------------
// attn_out = P @ V : per bnh, (512 q x 512 key) x (512 key x 64 hd)
// P loaded as f32 (from d_out attn region) and converted to bf16 in-register.
// Vt layout: [(bnh*64 + hd) * 512 + key]  (contiguous in key per lane)
// Result written bf16 into X3 columns [h*64, h*64+64).
// grid(4 qblocks, 256 bnh); wave handles 16 q rows x 64 hd.
// ---------------------------------------------------------------------------
__global__ __launch_bounds__(256) void attn_pv_kernel(
    const float* __restrict__ P, const bf16* __restrict__ Vt,
    bf16* __restrict__ X3)
{
    const int lane = threadIdx.x & 31;
    const int wave = threadIdx.x >> 5;
    const int half = lane >> 4;
    const int lr   = lane & 15;
    const int bnh = blockIdx.y;
    const int q0  = blockIdx.x * 128 + wave * 16;

    const float* abase = P + ((size_t)bnh * TT + q0 + lr) * TT + half * 16;
    const bf16*  vbase = Vt + (size_t)bnh * HD * TT + half * 16;

    v8f acc[4] = {};
    for (int k0 = 0; k0 < TT; k0 += 32) {
        const float* a0 = abase + k0;
        const float* a1 = abase + k0 + 8;
        float4 f0 = ((const float4*)a0)[0];
        float4 f1 = ((const float4*)a0)[1];
        float4 f2 = ((const float4*)a1)[0];
        float4 f3 = ((const float4*)a1)[1];
        ABu a;
        a.h[0] = cvt8(f0, f1);
        a.h[1] = cvt8(f2, f3);
#pragma unroll
        for (int nt = 0; nt < 4; ++nt) {
            v16bf bv = *(const v16bf*)(vbase + (size_t)(nt * 16 + lr) * TT + k0);
            acc[nt] = wmma_bf16(a.v, bv, acc[nt]);
        }
    }

    const int bnn = bnh >> 3, h = bnh & 7;
#pragma unroll
    for (int nt = 0; nt < 4; ++nt) {
        int hd = nt * 16 + lr;
#pragma unroll
        for (int r = 0; r < 8; ++r) {
            int m = bnn * TT + q0 + half * 8 + r;
            X3[(size_t)m * 1024 + h * HD + hd] = (bf16)acc[nt][r];
        }
    }
}

// ---------------------------------------------------------------------------
// Host-side launch
// ---------------------------------------------------------------------------
// workspace layout (bytes)
static const size_t OFF_A1  = 0;                       // 16,777,216 (bf16 16384x512)
static const size_t OFF_A2  = OFF_A1  + 16777216ull;
static const size_t OFF_QB  = OFF_A2  + 16777216ull;
static const size_t OFF_KB  = OFF_QB  + 16777216ull;
static const size_t OFF_VT  = OFF_KB  + 16777216ull;
static const size_t OFF_X3  = OFF_VT  + 16777216ull;   // 33,554,432 (bf16 16384x1024)
static const size_t OFF_Q0  = OFF_X3  + 33554432ull;   // 33,554,432 (f32)
static const size_t OFF_K0  = OFF_Q0  + 33554432ull;   // 33,554,432 (f32)
static const size_t OFF_WQ  = OFF_K0  + 33554432ull;   // 524,288 each
static const size_t OFF_WK  = OFF_WQ  + 524288ull;
static const size_t OFF_WV  = OFF_WK  + 524288ull;
static const size_t OFF_W1  = OFF_WV  + 524288ull;
static const size_t OFF_WC  = OFF_W1  + 524288ull;     // 1,048,576 (bf16 512x1024)
static const size_t OFF_PRT = OFF_WC  + 1048576ull;    // partials
static const size_t OFF_ST  = OFF_PRT + 8192ull;       // stats

static void run_gn(const float* X, bf16* Y, float* part, float* stats,
                   hipStream_t s) {
    gn_part_kernel<<<dim3(GNCH, BB * GG), 256, 0, s>>>(X, part);
    gn_stat_kernel<<<BB * GG, 32, 0, s>>>(part, stats);
    gn_apply_kernel<<<8192, 256, 0, s>>>(X, stats, Y);
}

extern "C" void kernel_launch(void* const* d_in, const int* in_sizes, int n_in,
                              void* d_out, int out_size, void* d_ws, size_t ws_size,
                              hipStream_t stream) {
    (void)in_sizes; (void)n_in; (void)out_size; (void)ws_size;

    const float* in_f    = (const float*)d_in[1];
    const float* in_f_kv = (const float*)d_in[2];
    const float* Wq = (const float*)d_in[3];
    const float* Wk = (const float*)d_in[4];
    const float* Wv = (const float*)d_in[5];
    const float* Wz = (const float*)d_in[6];
    const float* bz = (const float*)d_in[7];
    const float* W1 = (const float*)d_in[8];
    const float* b1 = (const float*)d_in[9];
    const float* W2 = (const float*)d_in[10];
    const float* b2 = (const float*)d_in[11];

    char* ws = (char*)d_ws;
    bf16* A1  = (bf16*)(ws + OFF_A1);
    bf16* A2  = (bf16*)(ws + OFF_A2);
    bf16* Qb  = (bf16*)(ws + OFF_QB);
    bf16* Kb  = (bf16*)(ws + OFF_KB);
    bf16* Vt  = (bf16*)(ws + OFF_VT);
    bf16* X3  = (bf16*)(ws + OFF_X3);
    float* Q0 = (float*)(ws + OFF_Q0);
    float* K0 = (float*)(ws + OFF_K0);
    bf16* Wqb = (bf16*)(ws + OFF_WQ);
    bf16* Wkb = (bf16*)(ws + OFF_WK);
    bf16* Wvb = (bf16*)(ws + OFF_WV);
    bf16* W1b = (bf16*)(ws + OFF_W1);
    bf16* Wcb = (bf16*)(ws + OFF_WC);
    float* part  = (float*)(ws + OFF_PRT);
    float* stats = (float*)(ws + OFF_ST);

    float* OUT = (float*)d_out;                    // (4,8,512,512) f32
    float* S   = OUT + (size_t)ME * DD;            // attn_weights (4,8,8,512,512)

    // 1) weights -> bf16 ([Wz|W2] concatenated for fused final GEMM)
    cvt_w_kernel<<<256, 256, 0, stream>>>(Wq, Wqb, 512, 0);
    cvt_w_kernel<<<256, 256, 0, stream>>>(Wk, Wkb, 512, 0);
    cvt_w_kernel<<<256, 256, 0, stream>>>(Wv, Wvb, 512, 0);
    cvt_w_kernel<<<256, 256, 0, stream>>>(W1, W1b, 512, 0);
    cvt_w_kernel<<<256, 256, 0, stream>>>(Wz, Wcb, 1024, 0);
    cvt_w_kernel<<<256, 256, 0, stream>>>(W2, Wcb, 1024, 512);

    // 2) GroupNorm inputs
    run_gn(in_f,    A1, part, stats, stream);
    run_gn(in_f_kv, A2, part, stats, stream);

    const dim3 ggrid(ME / 128, DD / 64);
    // 3) projections
    gemm_kernel<0><<<ggrid, 256, 0, stream>>>(A1, Wqb, 512, 512, Q0, nullptr,
                                              nullptr, nullptr, nullptr);
    gemm_kernel<0><<<ggrid, 256, 0, stream>>>(A2, Wkb, 512, 512, K0, nullptr,
                                              nullptr, nullptr, nullptr);
    gemm_kernel<1><<<ggrid, 256, 0, stream>>>(A2, Wvb, 512, 512, nullptr, Vt,
                                              nullptr, nullptr, nullptr);
    gemm_kernel<2><<<ggrid, 256, 0, stream>>>(A1, W1b, 512, 512, nullptr, X3,
                                              b1, nullptr, nullptr);

    // 4) GroupNorm on Q/K projections
    run_gn(Q0, Qb, part, stats, stream);
    run_gn(K0, Kb, part, stats, stream);

    // 5) attention
    attn_logits_kernel<<<dim3(32, 256), 256, 0, stream>>>(Qb, Kb, S);
    softmax_kernel<<<16384, 256, 0, stream>>>(S);
    attn_pv_kernel<<<dim3(4, 256), 256, 0, stream>>>(S, Vt, X3);

    // 6) fused residual GEMM: out = in_f + X3 @ [Wz|W2]^T + (bz + b2)
    gemm_kernel<3><<<ggrid, 256, 0, stream>>>(X3, Wcb, 1024, 1024, OUT, nullptr,
                                              bz, b2, in_f);
}
